// RWKV_52183852646461
// MI455X (gfx1250) — matrix operand order repaired and checked by reading, not compiled, not run on hardware
//
#include <hip/hip_runtime.h>
#include <hip/hip_bf16.h>

#define T_SEQ 2048
#define DIM   1024
#define HID   4096
#define VOCAB 50257
#define NLAYER 8
#define LN_EPS 1e-5f

// gfx1250 async memory->LDS staging for the bf16 A tiles (ASYNCcnt path).
#define USE_ASYNC_A 1

typedef __attribute__((ext_vector_type(8)))  float  v8f;
typedef __attribute__((ext_vector_type(8)))  __bf16 v8bf;
typedef __attribute__((ext_vector_type(16))) __bf16 v16bf;
typedef __attribute__((ext_vector_type(4)))  float  f4;

typedef __attribute__((address_space(3))) unsigned short lds_ushort;

union V16 { v16bf v; v8bf h[2]; };

__device__ __forceinline__ unsigned short f2bf(float f) {
  unsigned u = __float_as_uint(f);
  u += 0x7FFFu + ((u >> 16) & 1u);      // round-to-nearest-even
  return (unsigned short)(u >> 16);
}

// ---------------------------------------------------------------------------
// LayerNorm, block per token row. tokens!=nullptr -> fused embedding gather.
// out_bf!=nullptr -> emit bf16 (GEMM A operand), else fp32.
// ---------------------------------------------------------------------------
__global__ __launch_bounds__(256)
void ln_kernel(const float* __restrict__ in, const int* __restrict__ tokens,
               const float* __restrict__ emb,
               const float* __restrict__ w, const float* __restrict__ b,
               float* __restrict__ out, unsigned short* __restrict__ out_bf) {
  __shared__ float red[256];
  const int t = blockIdx.x;
  const float* row = tokens ? (emb + (size_t)tokens[t] * DIM)
                            : (in + (size_t)t * DIM);
  float vals[4];
  float s = 0.f;
#pragma unroll
  for (int j = 0; j < 4; ++j) { vals[j] = row[threadIdx.x + j * 256]; s += vals[j]; }
  red[threadIdx.x] = s; __syncthreads();
  for (int off = 128; off > 0; off >>= 1) {
    if (threadIdx.x < off) red[threadIdx.x] += red[threadIdx.x + off];
    __syncthreads();
  }
  float mean = red[0] * (1.0f / DIM);
  __syncthreads();
  float sq = 0.f;
#pragma unroll
  for (int j = 0; j < 4; ++j) { float d = vals[j] - mean; sq += d * d; }
  red[threadIdx.x] = sq; __syncthreads();
  for (int off = 128; off > 0; off >>= 1) {
    if (threadIdx.x < off) red[threadIdx.x] += red[threadIdx.x + off];
    __syncthreads();
  }
  float inv = rsqrtf(red[0] * (1.0f / DIM) + LN_EPS);
#pragma unroll
  for (int j = 0; j < 4; ++j) {
    int i = threadIdx.x + j * 256;
    float v = (vals[j] - mean) * inv * w[i] + b[i];
    if (out_bf) out_bf[(size_t)t * DIM + i] = f2bf(v);
    else        out[(size_t)t * DIM + i]    = v;
  }
}

// out[t] = x[t]*m + x[t-1]*(1-m) as bf16; grid = (DIM/256, T)
__global__ __launch_bounds__(256)
void mix_kernel(const float* __restrict__ x, const float* __restrict__ mix,
                unsigned short* __restrict__ out) {
  const int d = blockIdx.x * 256 + threadIdx.x;
  const int t = blockIdx.y;
  const size_t idx = (size_t)t * DIM + d;
  float m = mix[d];
  float prev = (t > 0) ? x[idx - DIM] : 0.0f;
  out[idx] = f2bf(x[idx] * m + prev * (1.0f - m));
}

// out = a*b as bf16; grid = (DIM/256, T)
__global__ __launch_bounds__(256)
void ew_mul_kernel(const float* __restrict__ a, const float* __restrict__ b,
                   unsigned short* __restrict__ out) {
  const size_t idx = (size_t)blockIdx.y * DIM + blockIdx.x * 256 + threadIdx.x;
  out[idx] = f2bf(a[idx] * b[idx]);
}

// ---------------------------------------------------------------------------
// Numerically-stable WKV recurrence: one thread per channel, sequential in T.
// ---------------------------------------------------------------------------
__global__ __launch_bounds__(256)
void wkv_kernel(const float* __restrict__ td, const float* __restrict__ tf,
                const float* __restrict__ k, const float* __restrict__ v,
                float* __restrict__ y) {
  int d = blockIdx.x * 256 + threadIdx.x;
  if (d >= DIM) return;
  float w = -__expf(td[d]);
  float u = tf[d];
  float aa = 0.f, bb = 0.f, pp = -1e30f;
  for (int t = 0; t < T_SEQ; ++t) {
    float kt = k[(size_t)t * DIM + d], vt = v[(size_t)t * DIM + d];
    float ww = u + kt;
    float qq = fmaxf(pp, ww);
    float e1 = __expf(pp - qq), e2 = __expf(ww - qq);
    y[(size_t)t * DIM + d] = (e1 * aa + e2 * vt) / (e1 * bb + e2);
    float ww2 = pp + w;
    float qq2 = fmaxf(ww2, kt);
    e1 = __expf(ww2 - qq2); e2 = __expf(kt - qq2);
    aa = e1 * aa + e2 * vt;
    bb = e1 * bb + e2;
    pp = qq2;
  }
}

// ---------------------------------------------------------------------------
// bf16 WMMA GEMM, double-buffered software pipeline, ONE barrier per K-step:
//   A: [M,K] bf16 -> LDS via GLOBAL_LOAD_ASYNC_TO_LDS_B128 (ASYNCcnt)
//   B: [K,N] fp32 -> VGPRs (b128) -> bf16 -> LDS transposed [n][k]
// 64x64 tile / 128 threads (4 waves, each 2x2 of v_wmma_f32_16x16x32_bf16).
// Grid: x = M/64 (consecutive blocks share the same B slice -> L2 reuse).
// Barrier proof: storeX(it) rewrites buf[it&1], last read by compute(it-2);
// every wave passes barrier(it-1) only after its compute(it-2), so the single
// top-of-iteration barrier orders the rewrite. Async writes to buf[(it+1)&1]
// are issued after barrier(it), whose arrival implies compute(it-1) (the last
// reader of that buffer) has finished on all waves.
// EPI: 0 none, 1 sigmoid, 2 relu^2, 3 out=add+acc, 4 out=add+mul*acc
// ---------------------------------------------------------------------------
template <int EPI, bool OBF, bool BCHK>
__global__ __launch_bounds__(128)
void gemm_bf16_wmma(const unsigned short* __restrict__ A,
                    const float* __restrict__ B, void* __restrict__ Cv,
                    const float* __restrict__ addv, const float* __restrict__ mulv,
                    int M, int N, int K) {
  constexpr int LDT = 40;  // halves per LDS row (80B; 16B-aligned vectors)
  __shared__ __align__(16) unsigned short As[2][64 * LDT];  // [m][k]
  __shared__ __align__(16) unsigned short Bs[2][64 * LDT];  // [n][k]

  const int tid  = threadIdx.x;
  const int lane = tid & 31;
  const int wave = tid >> 5;
  const int wm = wave >> 1, wn = wave & 1;
  const int m0 = blockIdx.x * 64;
  const int n0 = blockIdx.y * 64;
  const int lrow  = lane & 15;
  const int khalf = lane >> 4;

  v8f acc[2][2] = {};
  float bregs[16];
#if !USE_ASYNC_A
  uint4 aregs[2];
#endif

  // Per-thread A chunk coordinates (2 chunks of 8 halves)
  int arow[2], acol[2];
#pragma unroll
  for (int i = 0; i < 2; ++i) {
    int idx = tid + i * 128;
    arow[i] = idx >> 2;
    acol[i] = (idx & 3) * 8;
  }

  auto loadA = [&](int buf, int kt) {
#pragma unroll
    for (int i = 0; i < 2; ++i) {
      const unsigned short* g = &A[(size_t)(m0 + arow[i]) * K + kt + acol[i]];
#if USE_ASYNC_A
      lds_ushort* l = (lds_ushort*)&As[buf][arow[i] * LDT + acol[i]];
      asm volatile("global_load_async_to_lds_b128 %0, %1, off"
                   :: "v"(l), "v"(g) : "memory");
#else
      aregs[i] = *(const uint4*)g;
      (void)buf;
#endif
    }
  };
#if !USE_ASYNC_A
  auto storeA = [&](int buf) {
#pragma unroll
    for (int i = 0; i < 2; ++i)
      *(uint4*)&As[buf][arow[i] * LDT + acol[i]] = aregs[i];
  };
#endif

  auto loadB = [&](int kt) {
    if constexpr (BCHK) {
#pragma unroll
      for (int i = 0; i < 16; ++i) {
        int idx = tid + i * 128;
        int r = idx >> 6, c = idx & 63;
        bregs[i] = (n0 + c < N) ? B[(size_t)(kt + r) * N + (n0 + c)] : 0.0f;
      }
    } else {
#pragma unroll
      for (int i = 0; i < 4; ++i) {
        int idx = tid + i * 128;
        int r = idx >> 4, c4 = (idx & 15) * 4;
        *(f4*)&bregs[i * 4] = *(const f4*)&B[(size_t)(kt + r) * N + (n0 + c4)];
      }
    }
  };
  auto storeB = [&](int buf) {
    if constexpr (BCHK) {
#pragma unroll
      for (int i = 0; i < 16; ++i) {
        int idx = tid + i * 128;
        int r = idx >> 6, c = idx & 63;
        Bs[buf][c * LDT + r] = f2bf(bregs[i]);
      }
    } else {
#pragma unroll
      for (int i = 0; i < 4; ++i) {
        int idx = tid + i * 128;
        int r = idx >> 4, c4 = (idx & 15) * 4;
#pragma unroll
        for (int j = 0; j < 4; ++j)
          Bs[buf][(c4 + j) * LDT + r] = f2bf(bregs[i * 4 + j]);
      }
    }
  };

  const int nk = K >> 5;
  loadA(0, 0);
  loadB(0);

  for (int it = 0; it < nk; ++it) {
    const int cur = it & 1;
    storeB(cur);
#if !USE_ASYNC_A
    storeA(cur);
#else
    asm volatile("s_wait_asynccnt 0x0" ::: "memory");
#endif
    __syncthreads();   // single barrier per K-step (see proof above)

    if (it + 1 < nk) {          // issue next tile's loads under the WMMAs
      loadA((it + 1) & 1, (it + 1) * 32);
      loadB((it + 1) * 32);
    }

    const unsigned short* as = As[cur];
    const unsigned short* bs = Bs[cur];
#pragma unroll
    for (int ti = 0; ti < 2; ++ti) {
      // A frag (wave32): lanes<16 K{0..7,16..23}, lanes>=16 K{8..15,24..31}
      V16 a;
      int ar = wm * 32 + ti * 16 + lrow;
      a.h[0] = *(const v8bf*)&as[ar * LDT + khalf * 8];
      a.h[1] = *(const v8bf*)&as[ar * LDT + 16 + khalf * 8];
#pragma unroll
      for (int tj = 0; tj < 2; ++tj) {
        // B frag: lanes<16 K{0..7,16..23}, lanes>=16 K{8..15,24..31}; col = lane&15
        V16 b;
        int bn = wn * 32 + tj * 16 + lrow;
        b.h[0] = *(const v8bf*)&bs[bn * LDT + khalf * 8];
        b.h[1] = *(const v8bf*)&bs[bn * LDT + 16 + khalf * 8];
        acc[ti][tj] = __builtin_amdgcn_wmma_f32_16x16x32_bf16(
            false, a.v, false, b.v, (short)0, acc[ti][tj], false, false);
      }
    }
  }

  // Epilogue + store. C/D layout: VGPR r -> M = r + 8*(lane>>4), N = lane&15.
  float* Cf = (float*)Cv;
  unsigned short* Cb = (unsigned short*)Cv;
#pragma unroll
  for (int ti = 0; ti < 2; ++ti)
#pragma unroll
    for (int tj = 0; tj < 2; ++tj) {
      int n = n0 + wn * 32 + tj * 16 + lrow;
      if (BCHK && n >= N) continue;
#pragma unroll
      for (int r = 0; r < 8; ++r) {
        int m = m0 + wm * 32 + ti * 16 + khalf * 8 + r;
        size_t off = (size_t)m * N + n;
        float v = acc[ti][tj][r];
        if (EPI == 1)      v = 1.0f / (1.0f + __expf(-v));
        else if (EPI == 2) { v = v > 0.f ? v : 0.f; v = v * v; }
        else if (EPI == 3) v = addv[off] + v;
        else if (EPI == 4) v = addv[off] + mulv[off] * v;
        if (OBF) Cb[off] = f2bf(v);
        else     Cf[off] = v;
      }
    }
}

// ---------------------------------------------------------------------------
extern "C" void kernel_launch(void* const* d_in, const int* in_sizes, int n_in,
                              void* d_out, int out_size, void* d_ws, size_t ws_size,
                              hipStream_t stream) {
  const int*   tokens   = (const int*)  d_in[0];
  const float* emb      = (const float*)d_in[1];
  const float* tm_decay = (const float*)d_in[2];
  const float* tm_first = (const float*)d_in[3];
  const float* tm_mix_k = (const float*)d_in[4];
  const float* tm_mix_v = (const float*)d_in[5];
  const float* tm_mix_r = (const float*)d_in[6];
  const float* tm_Wk    = (const float*)d_in[7];
  const float* tm_Wv    = (const float*)d_in[8];
  const float* tm_Wr    = (const float*)d_in[9];
  const float* tm_Wo    = (const float*)d_in[10];
  const float* cm_mix_k = (const float*)d_in[11];
  const float* cm_mix_r = (const float*)d_in[12];
  const float* cm_Wk    = (const float*)d_in[13];
  const float* cm_Wv    = (const float*)d_in[14];
  const float* cm_Wr    = (const float*)d_in[15];
  const float* ln0_w    = (const float*)d_in[16];
  const float* ln0_b    = (const float*)d_in[17];
  const float* ln1_w    = (const float*)d_in[18];
  const float* ln1_b    = (const float*)d_in[19];
  const float* ln2_w    = (const float*)d_in[20];
  const float* ln2_b    = (const float*)d_in[21];
  const float* lno_w    = (const float*)d_in[22];
  const float* lno_b    = (const float*)d_in[23];
  const float* head_W   = (const float*)d_in[24];
  float* out = (float*)d_out;

  const size_t TD = (size_t)T_SEQ * DIM;
  const size_t TH = (size_t)T_SEQ * HID;
  char* wsp = (char*)d_ws;
  float* x   = (float*)wsp; wsp += TD * 4;   // residual stream
  float* xn  = (float*)wsp; wsp += TD * 4;   // ln2 output
  float* kb  = (float*)wsp; wsp += TD * 4;
  float* vb  = (float*)wsp; wsp += TD * 4;
  float* rb  = (float*)wsp; wsp += TD * 4;
  float* yb  = (float*)wsp; wsp += TD * 4;
  float* rrb = (float*)wsp; wsp += TD * 4;
  unsigned short* bufA = (unsigned short*)wsp; wsp += TD * 2;  // bf16 GEMM-A
  unsigned short* kkbf = (unsigned short*)wsp; wsp += TH * 2;  // bf16 relu^2
  unsigned short* xnbf = (unsigned short*)wsp; wsp += TD * 2;  // bf16 head A

  const dim3 blk(128);
  const dim3 ewGrid(DIM / 256, T_SEQ);
  auto gg = [](int N) { return dim3(T_SEQ / 64, (unsigned)((N + 63) / 64)); };

  // x = ln0(emb[tokens])
  ln_kernel<<<T_SEQ, 256, 0, stream>>>(nullptr, tokens, emb, ln0_w, ln0_b, x, nullptr);

  for (int l = 0; l < NLAYER; ++l) {
    const float* Wk  = tm_Wk + (size_t)l * DIM * DIM;
    const float* Wv  = tm_Wv + (size_t)l * DIM * DIM;
    const float* Wr  = tm_Wr + (size_t)l * DIM * DIM;
    const float* Wo  = tm_Wo + (size_t)l * DIM * DIM;
    const float* CWk = cm_Wk + (size_t)l * DIM * HID;
    const float* CWv = cm_Wv + (size_t)l * HID * DIM;
    const float* CWr = cm_Wr + (size_t)l * DIM * DIM;

    // x = ln1(x)  (quirk: ln1 output IS the residual stream)
    ln_kernel<<<T_SEQ, 256, 0, stream>>>(x, nullptr, nullptr,
                                         ln1_w + l * DIM, ln1_b + l * DIM, x, nullptr);
    // k = mix_k(x) @ Wk
    mix_kernel<<<ewGrid, 256, 0, stream>>>(x, tm_mix_k + l * DIM, bufA);
    gemm_bf16_wmma<0, false, false><<<gg(DIM), blk, 0, stream>>>(
        bufA, Wk, kb, nullptr, nullptr, T_SEQ, DIM, DIM);
    // v = mix_v(x) @ Wv
    mix_kernel<<<ewGrid, 256, 0, stream>>>(x, tm_mix_v + l * DIM, bufA);
    gemm_bf16_wmma<0, false, false><<<gg(DIM), blk, 0, stream>>>(
        bufA, Wv, vb, nullptr, nullptr, T_SEQ, DIM, DIM);
    // r = sigmoid(mix_r(x) @ Wr)
    mix_kernel<<<ewGrid, 256, 0, stream>>>(x, tm_mix_r + l * DIM, bufA);
    gemm_bf16_wmma<1, false, false><<<gg(DIM), blk, 0, stream>>>(
        bufA, Wr, rb, nullptr, nullptr, T_SEQ, DIM, DIM);
    // y = wkv(-exp(td), tf, k, v)
    wkv_kernel<<<DIM / 256, 256, 0, stream>>>(tm_decay + l * DIM, tm_first + l * DIM,
                                              kb, vb, yb);
    // x = x + (r*y) @ Wo
    ew_mul_kernel<<<ewGrid, 256, 0, stream>>>(rb, yb, bufA);
    gemm_bf16_wmma<3, false, false><<<gg(DIM), blk, 0, stream>>>(
        bufA, Wo, x, x, nullptr, T_SEQ, DIM, DIM);
    // channel-mix
    ln_kernel<<<T_SEQ, 256, 0, stream>>>(x, nullptr, nullptr,
                                         ln2_w + l * DIM, ln2_b + l * DIM, xn, nullptr);
    mix_kernel<<<ewGrid, 256, 0, stream>>>(xn, cm_mix_k + l * DIM, bufA);
    gemm_bf16_wmma<2, true, false><<<gg(HID), blk, 0, stream>>>(
        bufA, CWk, kkbf, nullptr, nullptr, T_SEQ, HID, DIM);
    mix_kernel<<<ewGrid, 256, 0, stream>>>(xn, cm_mix_r + l * DIM, bufA);
    gemm_bf16_wmma<1, false, false><<<gg(DIM), blk, 0, stream>>>(
        bufA, CWr, rrb, nullptr, nullptr, T_SEQ, DIM, DIM);
    // x = x + rr * (kk @ CWv)
    gemm_bf16_wmma<4, false, false><<<gg(DIM), blk, 0, stream>>>(
        kkbf, CWv, x, x, rrb, T_SEQ, DIM, HID);
  }

  // logits = ln_out(x) @ head_W   (N = 50257 -> bounds-checked path)
  ln_kernel<<<T_SEQ, 256, 0, stream>>>(x, nullptr, nullptr, lno_w, lno_b,
                                       nullptr, xnbf);
  gemm_bf16_wmma<0, false, true><<<gg(VOCAB), blk, 0, stream>>>(
      xnbf, head_W, out, nullptr, nullptr, T_SEQ, VOCAB, DIM);
}
